// GaussianImage_Cholesky_83726092469098
// MI455X (gfx1250) — compile-verified, matrix operand order checked by reading, not run on hardware
//
#include <hip/hip_runtime.h>
#include <hip/hip_bf16.h>
#include <math.h>

typedef __attribute__((ext_vector_type(16))) _Float16 v16h;
typedef __attribute__((ext_vector_type(8)))  _Float16 v8h;
typedef __attribute__((ext_vector_type(8)))  float    v8f;
typedef __attribute__((ext_vector_type(4)))  float    v4f;

#define NPTS   2048
#define WIDTH  128
#define HEIGHT 128
#define TFR    2
#define TN     (TFR * NPTS)

// ---------------------------------------------------------------------------
// Prep kernel: evaluate polynomial trajectories, conic inversion, sigmoids.
// One thread per (t, n).  Also writes channel-major sigmoid colors when t==0.
// ws layout (floats):
//   [0*TN) a  | [1*TN) b | [2*TN) c | [3*TN) op | [4*TN) mx | [5*TN) my
//   [6*TN)  colors[3][NPTS]
// ---------------------------------------------------------------------------
__global__ __launch_bounds__(256) void prep_kernel(
    const float* __restrict__ xyz,   // [3][N][2]
    const float* __restrict__ chol,  // [3][N][3]
    const float* __restrict__ opc,   // [3][N][1]
    const float* __restrict__ fdc,   // [N][3]
    const float* __restrict__ tvals, // [T]
    float* __restrict__ ws)
{
    int idx = blockIdx.x * blockDim.x + threadIdx.x;
    if (idx >= TN) return;
    int t = idx / NPTS;
    int n = idx - t * NPTS;

    float tv  = tvals[t];
    float tp1 = tv, tp2 = tv * tv;

    // xyz -> tanh -> pixel center
    float xr = xyz[(0 * NPTS + n) * 2 + 0]
             + tp1 * xyz[(1 * NPTS + n) * 2 + 0]
             + tp2 * xyz[(2 * NPTS + n) * 2 + 0];
    float yr = xyz[(0 * NPTS + n) * 2 + 1]
             + tp1 * xyz[(1 * NPTS + n) * 2 + 1]
             + tp2 * xyz[(2 * NPTS + n) * 2 + 1];
    float mx = 0.5f * (tanhf(xr) + 1.0f) * (float)WIDTH;
    float my = 0.5f * (tanhf(yr) + 1.0f) * (float)HEIGHT;

    // cholesky -> Sigma -> conic (inverse covariance)
    float l0 = 0.5f + chol[(0 * NPTS + n) * 3 + 0]
                    + tp1 * chol[(1 * NPTS + n) * 3 + 0]
                    + tp2 * chol[(2 * NPTS + n) * 3 + 0];
    float l1 =        chol[(0 * NPTS + n) * 3 + 1]
                    + tp1 * chol[(1 * NPTS + n) * 3 + 1]
                    + tp2 * chol[(2 * NPTS + n) * 3 + 1];
    float l2 = 0.5f + chol[(0 * NPTS + n) * 3 + 2]
                    + tp1 * chol[(1 * NPTS + n) * 3 + 2]
                    + tp2 * chol[(2 * NPTS + n) * 3 + 2];
    float s11 = l0 * l0;
    float s12 = l0 * l1;
    float s22 = l1 * l1 + l2 * l2;
    float det = s11 * s22 - s12 * s12;
    float inv = 1.0f / det;

    // opacity
    float ov = opc[0 * NPTS + n] + tp1 * opc[1 * NPTS + n] + tp2 * opc[2 * NPTS + n];
    float op = 1.0f / (1.0f + __expf(-ov));

    ws[0 * TN + idx] = s22 * inv;    // a
    ws[1 * TN + idx] = -s12 * inv;   // b
    ws[2 * TN + idx] = s11 * inv;    // c
    ws[3 * TN + idx] = op;
    ws[4 * TN + idx] = mx;
    ws[5 * TN + idx] = my;

    if (t == 0) {
        float* col = ws + 6 * TN;    // [3][NPTS]
        #pragma unroll
        for (int ch = 0; ch < 3; ++ch)
            col[ch * NPTS + n] = 1.0f / (1.0f + __expf(-fdc[n * 3 + ch]));
    }
}

// ---------------------------------------------------------------------------
// Render kernel: one block per (frame, image row).  8 waves/block, each wave
// owns 16 consecutive pixels.  Loop over points in K=32 chunks; weights are
// computed in registers, packed to f16, and accumulated with
// v_wmma_f32_16x16x32_f16 against the (row-clamped) 16xK color matrix.
// Rows 3..15 of A are never zeroed: row m of D depends only on row m of A,
// and we only store rows 0..2.
// ---------------------------------------------------------------------------
__global__ __launch_bounds__(256) void render_kernel(
    const float* __restrict__ ws,
    float* __restrict__ out)   // [T][3][H][W]
{
    __shared__ float     s_a [NPTS];
    __shared__ float     s_b [NPTS];
    __shared__ float     s_c [NPTS];
    __shared__ float     s_op[NPTS];
    __shared__ float     s_mx[NPTS];
    __shared__ float     s_my[NPTS];
    __shared__ _Float16  s_col[3][NPTS];   // row stride 4096 B -> 16B-aligned runs

    const int bid  = blockIdx.x;
    const int t    = bid / HEIGHT;
    const int h    = bid - t * HEIGHT;
    const int tid  = threadIdx.x;
    const int base = t * NPTS;

    // Stage this frame's point parameters and the colors into LDS.
    for (int i = tid; i < NPTS; i += 256) {
        s_a [i] = ws[0 * TN + base + i];
        s_b [i] = ws[1 * TN + base + i];
        s_c [i] = ws[2 * TN + base + i];
        s_op[i] = ws[3 * TN + base + i];
        s_mx[i] = ws[4 * TN + base + i];
        s_my[i] = ws[5 * TN + base + i];
        const float* col = ws + 6 * TN;
        s_col[0][i] = (_Float16)col[0 * NPTS + i];
        s_col[1][i] = (_Float16)col[1 * NPTS + i];
        s_col[2][i] = (_Float16)col[2 * NPTS + i];
    }
    __syncthreads();

    const int wave = tid >> 5;          // 8 waves -> 8 x 16-pixel tiles = row
    const int lane = tid & 31;
    const int col  = lane & 15;         // pixel within tile / matrix row
    const int hi   = lane >> 4;         // lane-half selects K sub-range
    const int x    = wave * 16 + col;

    const float px = (float)x + 0.5f;
    const float py = (float)h + 0.5f;
    const int   m2 = (col < 3) ? col : 0;   // clamped A row (rows 3..15 = junk)

    v8f acc = {};

    #pragma unroll 2
    for (int kb = 0; kb < NPTS; kb += 32) {
        const int p0 = kb + hi * 8;     // first 8-point run (K 0..7 / 8..15)
        // ---- A: two 128-bit LDS loads of colors for this lane's row -------
        v8h ca0 = *(const v8h*)&s_col[m2][p0];
        v8h ca1 = *(const v8h*)&s_col[m2][p0 + 16];
        v16h av = __builtin_shufflevector(ca0, ca1,
                    0, 1, 2, 3, 4, 5, 6, 7, 8, 9, 10, 11, 12, 13, 14, 15);

        // ---- B: weights for 2 runs of 8 points, vector LDS loads ----------
        v16h bv;
        #pragma unroll
        for (int r = 0; r < 2; ++r) {
            const int pr = p0 + r * 16;
            float pa[8], pb[8], pc[8], po[8], pmx[8], pmy[8];
            *(v4f*)&pa [0] = *(const v4f*)&s_a [pr];
            *(v4f*)&pa [4] = *(const v4f*)&s_a [pr + 4];
            *(v4f*)&pb [0] = *(const v4f*)&s_b [pr];
            *(v4f*)&pb [4] = *(const v4f*)&s_b [pr + 4];
            *(v4f*)&pc [0] = *(const v4f*)&s_c [pr];
            *(v4f*)&pc [4] = *(const v4f*)&s_c [pr + 4];
            *(v4f*)&po [0] = *(const v4f*)&s_op[pr];
            *(v4f*)&po [4] = *(const v4f*)&s_op[pr + 4];
            *(v4f*)&pmx[0] = *(const v4f*)&s_mx[pr];
            *(v4f*)&pmx[4] = *(const v4f*)&s_mx[pr + 4];
            *(v4f*)&pmy[0] = *(const v4f*)&s_my[pr];
            *(v4f*)&pmy[4] = *(const v4f*)&s_my[pr + 4];
            #pragma unroll
            for (int j = 0; j < 8; ++j) {
                float dx = pmx[j] - px;
                float dy = pmy[j] - py;
                float q  = pa[j] * dx * dx
                         + 2.0f * pb[j] * dx * dy
                         + pc[j] * dy * dy;
                float w  = __expf(-0.5f * q) * po[j];
                bv[r * 8 + j] = (_Float16)w;
            }
        }

        acc = __builtin_amdgcn_wmma_f32_16x16x32_f16(
            /*neg_a=*/false, av, /*neg_b=*/false, bv,
            /*c_mod=*/(short)0, acc, /*reuse_a=*/false, /*reuse_b=*/false);
    }

    // C/D layout: VGPR r of lanes 0..15 holds row r (channel), col = lane.
    // Channels 3..15 hold junk from the clamped A rows; only rows 0..2 stored.
    if (hi == 0) {
        #pragma unroll
        for (int ch = 0; ch < 3; ++ch) {
            float v = fminf(fmaxf(acc[ch], 0.0f), 1.0f);
            out[((t * 3 + ch) * HEIGHT + h) * WIDTH + x] = v;
        }
    }
}

// ---------------------------------------------------------------------------
extern "C" void kernel_launch(void* const* d_in, const int* in_sizes, int n_in,
                              void* d_out, int out_size, void* d_ws, size_t ws_size,
                              hipStream_t stream) {
    (void)in_sizes; (void)n_in; (void)out_size; (void)ws_size;
    const float* xyz   = (const float*)d_in[0];
    const float* chol  = (const float*)d_in[1];
    const float* opc   = (const float*)d_in[2];
    const float* fdc   = (const float*)d_in[3];
    const float* tvals = (const float*)d_in[4];
    float* ws  = (float*)d_ws;
    float* out = (float*)d_out;

    prep_kernel<<<(TN + 255) / 256, 256, 0, stream>>>(xyz, chol, opc, fdc, tvals, ws);
    render_kernel<<<TFR * HEIGHT, 256, 0, stream>>>(ws, out);
}
